// CompGraphConv_60559038873715
// MI455X (gfx1250) — matrix operand order repaired and checked by reading, not compiled, not run on hardware
//
#include <hip/hip_runtime.h>
#include <hip/hip_bf16.h>

#define Nn     50000
#define Ee     800000
#define D_IN   100
#define D_OUT  100
#define Rr     200
#define Bb     50
#define BN_EPS 1e-5f

typedef _Float16 v16h __attribute__((ext_vector_type(16)));
typedef _Float16 v8h  __attribute__((ext_vector_type(8)));
typedef float    v8f  __attribute__((ext_vector_type(8)));

// ---------------------------------------------------------------------------
// Pack fp32 matrix into padded fp16 buffer (handles optional transpose):
//   outh[r*Kp + c] = (r < M && c < K) ? (f16) in[r*sN + c*sK] : 0
// ---------------------------------------------------------------------------
__global__ __launch_bounds__(256) void pack_f16_kernel(
    const float* __restrict__ in, _Float16* __restrict__ outh,
    int M, int K, int Mpad, int Kp, int sN, int sK)
{
    const size_t i = (size_t)blockIdx.x * blockDim.x + threadIdx.x;
    const size_t total = (size_t)Mpad * Kp;
    if (i >= total) return;
    const int r = (int)(i / Kp);
    const int c = (int)(i % Kp);
    float v = 0.0f;
    if (r < M && c < K) v = in[(size_t)r * sN + (size_t)c * sK];
    outh[i] = (_Float16)v;
}

// ---------------------------------------------------------------------------
// WMMA GEMM on packed fp16 operands:
//   out[m,n] = sum_k Ah[m,k] * Bh[n,k] + bias[n]
//   Ah : [Mpad, Kp] fp16 row-major (Mpad % 16 == 0, Kp % 32 == 0, zero padded)
//   Bh : [Npad, Kp] fp16 row-major (Npad % 16 == 0, zero padded)
// One wave per 16x16 output tile; contiguous b128 fragment loads, no
// predication around the WMMA (EXEC all-ones as the ISA requires).
// ---------------------------------------------------------------------------
__global__ __launch_bounds__(256) void wmma_gemm_f16(
    const _Float16* __restrict__ Ah, const _Float16* __restrict__ Bh,
    const float* __restrict__ bias, float* __restrict__ out,
    int M, int Nout, int Kp, int totalWaves)
{
    const int wid = (int)((blockIdx.x * blockDim.x + threadIdx.x) >> 5);
    if (wid >= totalWaves) return;                 // wave-uniform exit
    const int tilesN = (Nout + 15) >> 4;
    const int tileM  = wid / tilesN;
    const int tileN  = wid % tilesN;
    const int lane   = (int)(threadIdx.x & 31);
    const int hi     = lane >> 4;                  // half-wave select
    const int l15    = lane & 15;

    // A fragment: row = tileM*16 + l15 ; halves [16*hi .. 16*hi+15]
    const _Float16* ap = Ah + (size_t)(tileM * 16 + l15) * Kp + 16 * hi;
    // B fragment: col = tileN*16 + l15 ; halves [16*hi .. 16*hi+15]
    const _Float16* bp = Bh + (size_t)(tileN * 16 + l15) * Kp + 16 * hi;

    v8f acc = {};
    for (int k0 = 0; k0 < Kp; k0 += 32) {
        const v8h alo = *(const v8h*)(ap + k0);        // 16B load
        const v8h ahi = *(const v8h*)(ap + k0 + 8);    // 16B load
        const v16h a = __builtin_shufflevector(alo, ahi,
            0, 1, 2, 3, 4, 5, 6, 7, 8, 9, 10, 11, 12, 13, 14, 15);
        const v16h b = *(const v16h*)(bp + k0);        // 32B load
        acc = __builtin_amdgcn_wmma_f32_16x16x32_f16(
                  false, a, false, b, (short)0, acc, false, false);
    }

    const int nCol = tileN * 16 + l15;
    const float bv = (bias != nullptr && nCol < Nout) ? bias[nCol] : 0.0f;
    #pragma unroll
    for (int v = 0; v < 8; ++v) {
        const int m = tileM * 16 + v + 8 * hi;     // C/D layout: VGPR v -> row v (+8 upper)
        if (m < M && nCol < Nout)
            out[(size_t)m * Nout + nCol] = acc[v] + bv;
    }
}

// ---------------------------------------------------------------------------
// Order-preserving float<->uint map for atomic segment-max
// ---------------------------------------------------------------------------
__device__ __forceinline__ unsigned f2ord(float f) {
    unsigned u = __float_as_uint(f);
    return (u & 0x80000000u) ? ~u : (u | 0x80000000u);
}
__device__ __forceinline__ float ord2f(unsigned u) {
    unsigned b = (u & 0x80000000u) ? (u & 0x7FFFFFFFu) : ~u;
    return __uint_as_float(b);
}

// att[e] = sum_d k[src[e],d] * relw[etype[e],d] * q[dst[e],d]   (wave32 per edge)
__global__ __launch_bounds__(256) void edge_att_kernel(
    const float* __restrict__ k, const float* __restrict__ q,
    const float* __restrict__ relw,
    const int* __restrict__ src, const int* __restrict__ dst,
    const int* __restrict__ etype, float* __restrict__ att, int E)
{
    const int e    = (int)((blockIdx.x * blockDim.x + threadIdx.x) >> 5);
    const int lane = (int)(threadIdx.x & 31);
    if (e >= E) return;
    const int s = src[e], d = dst[e], t = etype[e];
    const float* kp = k    + (size_t)s * D_OUT;
    const float* qp = q    + (size_t)d * D_OUT;
    const float* wp = relw + (size_t)t * D_OUT;
    float acc = 0.0f;
    for (int i = lane; i < D_OUT; i += 32)
        acc += kp[i] * wp[i] * qp[i];
    #pragma unroll
    for (int off = 16; off > 0; off >>= 1)
        acc += __shfl_xor(acc, off, 32);
    if (lane == 0) att[e] = acc;
}

__global__ __launch_bounds__(256) void edge_max_kernel(
    const float* __restrict__ att, const int* __restrict__ dst,
    unsigned* __restrict__ segmax, int E)
{
    const int e = (int)(blockIdx.x * blockDim.x + threadIdx.x);
    if (e >= E) return;
    atomicMax(&segmax[dst[e]], f2ord(att[e]));
}

__global__ __launch_bounds__(256) void edge_exp_kernel(
    const float* __restrict__ att, const int* __restrict__ dst,
    const unsigned* __restrict__ segmax, float* __restrict__ ex,
    float* __restrict__ denom, int E)
{
    const int e = (int)(blockIdx.x * blockDim.x + threadIdx.x);
    if (e >= E) return;
    const int d = dst[e];
    const float v = expf(att[e] - ord2f(segmax[d]));
    ex[e] = v;
    atomicAdd(&denom[d], v);
}

// final[dst,d] += (ex/denom[dst]) * v[src,d]   (wave32 per edge)
__global__ __launch_bounds__(256) void edge_scatter_kernel(
    const float* __restrict__ ex, const float* __restrict__ denom,
    const float* __restrict__ vfeat,
    const int* __restrict__ src, const int* __restrict__ dst,
    float* __restrict__ finalv, int E)
{
    const int e    = (int)((blockIdx.x * blockDim.x + threadIdx.x) >> 5);
    const int lane = (int)(threadIdx.x & 31);
    if (e >= E) return;
    const int s = src[e], d = dst[e];
    const float w = ex[e] / denom[d];
    const float* vp = vfeat  + (size_t)s * D_OUT;
    float*       fp = finalv + (size_t)d * D_OUT;
    for (int i = lane; i < D_OUT; i += 32)
        atomicAdd(&fp[i], w * vp[i]);
}

// pre = a*s + (1-a)*final   (written in-place into `finalv`)
__global__ __launch_bounds__(256) void combine_kernel(
    const float* __restrict__ sbuf, float* __restrict__ finalv,
    const float* __restrict__ alpha, size_t total)
{
    const size_t i = (size_t)blockIdx.x * blockDim.x + threadIdx.x;
    if (i >= total) return;
    const float a = 1.0f / (1.0f + expf(-alpha[0]));
    finalv[i] = a * sbuf[i] + (1.0f - a) * finalv[i];
}

// one block per feature: mean/var over the node axis
__global__ __launch_bounds__(256) void bn_stats_kernel(
    const float* __restrict__ pre, float* __restrict__ stats)
{
    const int d = (int)blockIdx.x;
    float s = 0.0f, ss = 0.0f;
    for (int n = (int)threadIdx.x; n < Nn; n += 256) {
        const float x = pre[(size_t)n * D_OUT + d];
        s += x; ss += x * x;
    }
    __shared__ float sh0[256], sh1[256];
    sh0[threadIdx.x] = s; sh1[threadIdx.x] = ss;
    __syncthreads();
    for (int off = 128; off > 0; off >>= 1) {
        if ((int)threadIdx.x < off) {
            sh0[threadIdx.x] += sh0[threadIdx.x + off];
            sh1[threadIdx.x] += sh1[threadIdx.x + off];
        }
        __syncthreads();
    }
    if (threadIdx.x == 0) {
        const float mu  = sh0[0] / (float)Nn;
        const float var = sh1[0] / (float)Nn - mu * mu;
        stats[d]         = mu;
        stats[D_OUT + d] = var;
    }
}

__global__ __launch_bounds__(256) void bn_tanh_kernel(
    const float* __restrict__ pre, const float* __restrict__ stats,
    const float* __restrict__ gamma, const float* __restrict__ beta,
    float* __restrict__ out, size_t total)
{
    const size_t i = (size_t)blockIdx.x * blockDim.x + threadIdx.x;
    if (i >= total) return;
    const int d = (int)(i % D_OUT);
    const float mu  = stats[d];
    const float var = stats[D_OUT + d];
    const float x = (pre[i] - mu) * rsqrtf(var + BN_EPS) * gamma[d] + beta[d];
    out[i] = tanhf(x);
}

// ---------------------------------------------------------------------------
extern "C" void kernel_launch(void* const* d_in, const int* in_sizes, int n_in,
                              void* d_out, int out_size, void* d_ws, size_t ws_size,
                              hipStream_t stream)
{
    const float* n_in_feats = (const float*)d_in[0];
    const float* r_feats    = (const float*)d_in[1];
    const int*   src        = (const int*)  d_in[2];
    const int*   dst        = (const int*)  d_in[3];
    const int*   etype      = (const int*)  d_in[4];
    /* d_in[5] = norm : unused by the reference */
    const float* Ws_w = (const float*)d_in[6];  const float* Ws_b = (const float*)d_in[7];
    const float* Wk_w = (const float*)d_in[8];  const float* Wk_b = (const float*)d_in[9];
    const float* Wq_w = (const float*)d_in[10]; const float* Wq_b = (const float*)d_in[11];
    const float* Wv_w = (const float*)d_in[12]; const float* Wv_b = (const float*)d_in[13];
    const float* Wr_w = (const float*)d_in[14]; const float* Wr_b = (const float*)d_in[15];
    const float* relation_att = (const float*)d_in[16];   // [B, D_OUT]
    const float* w_comp       = (const float*)d_in[17];   // [R, B]
    const float* alpha        = (const float*)d_in[18];
    /* d_in[19] = loop_rel : its projected row is sliced off by the reference */
    const float* bn_gamma = (const float*)d_in[20];
    const float* bn_beta  = (const float*)d_in[21];

    float* out = (float*)d_out;

    // ---- padded geometry ----
    const int KpIn  = 128;   // D_IN=100  -> 128
    const int KpB   = 64;    // Bb=50     -> 64
    const int NoutP = 112;   // D_OUT=100 -> 112
    const int RrP   = 208;   // Rr=200    -> 208

    // ---- workspace layout ----
    const size_t ND = (size_t)Nn * D_OUT;
    float* ws    = (float*)d_ws;
    float* kbuf  = ws;                          // [N, D_OUT]
    float* qbuf  = kbuf + ND;                   // [N, D_OUT]
    float* vbuf  = qbuf + ND;                   // [N, D_OUT]
    float* sbuf  = vbuf + ND;                   // [N, D_OUT]
    float* fbuf  = sbuf + ND;                   // [N, D_OUT]  final -> pre (in place)
    float* relw  = fbuf + ND;                   // [R, D_OUT]
    float* att   = relw + (size_t)Rr * D_OUT;   // [E]
    float* ex    = att + Ee;                    // [E]
    float* denom = ex + Ee;                     // [N]
    unsigned* segmax = (unsigned*)(denom + Nn); // [N]
    float* stats = (float*)(segmax + Nn);       // [256]
    // fp16 packed region (starts 64B-aligned: all previous counts are *16 floats)
    _Float16* hbase   = (_Float16*)(stats + 256);
    _Float16* Xh      = hbase;                              // [Nn,   KpIn]
    _Float16* Wkh     = Xh   + (size_t)Nn  * KpIn;          // [NoutP,KpIn]
    _Float16* Wqh     = Wkh  + (size_t)NoutP * KpIn;
    _Float16* Wvh     = Wqh  + (size_t)NoutP * KpIn;
    _Float16* Wsh     = Wvh  + (size_t)NoutP * KpIn;
    _Float16* Wrh     = Wsh  + (size_t)NoutP * KpIn;
    _Float16* wcomph  = Wrh  + (size_t)NoutP * KpIn;        // [RrP, KpB]
    _Float16* relatth = wcomph + (size_t)RrP * KpB;         // [NoutP,KpB]  (transposed)
    _Float16* rfeath  = relatth + (size_t)NoutP * KpB;      // [RrP, KpIn]

    // ---- zero-init accumulators ----
    hipMemsetAsync(fbuf,   0, ND * sizeof(float),    stream);
    hipMemsetAsync(denom,  0, Nn * sizeof(float),    stream);
    hipMemsetAsync(segmax, 0, Nn * sizeof(unsigned), stream);

    // ---- pack fp32 -> padded fp16 ----
    auto packGrid = [](size_t n) { return (int)((n + 255) / 256); };
    pack_f16_kernel<<<packGrid((size_t)Nn * KpIn), 256, 0, stream>>>(
        n_in_feats, Xh, Nn, D_IN, Nn, KpIn, D_IN, 1);
    pack_f16_kernel<<<packGrid((size_t)NoutP * KpIn), 256, 0, stream>>>(
        Wk_w, Wkh, D_OUT, D_IN, NoutP, KpIn, D_IN, 1);
    pack_f16_kernel<<<packGrid((size_t)NoutP * KpIn), 256, 0, stream>>>(
        Wq_w, Wqh, D_OUT, D_IN, NoutP, KpIn, D_IN, 1);
    pack_f16_kernel<<<packGrid((size_t)NoutP * KpIn), 256, 0, stream>>>(
        Wv_w, Wvh, D_OUT, D_IN, NoutP, KpIn, D_IN, 1);
    pack_f16_kernel<<<packGrid((size_t)NoutP * KpIn), 256, 0, stream>>>(
        Ws_w, Wsh, D_OUT, D_IN, NoutP, KpIn, D_IN, 1);
    pack_f16_kernel<<<packGrid((size_t)NoutP * KpIn), 256, 0, stream>>>(
        Wr_w, Wrh, D_OUT, D_IN, NoutP, KpIn, D_IN, 1);
    pack_f16_kernel<<<packGrid((size_t)RrP * KpB), 256, 0, stream>>>(
        w_comp, wcomph, Rr, Bb, RrP, KpB, Bb, 1);
    // relation_att transposed: B(n,k) = relation_att[k*D_OUT + n]
    pack_f16_kernel<<<packGrid((size_t)NoutP * KpB), 256, 0, stream>>>(
        relation_att, relatth, D_OUT, Bb, NoutP, KpB, 1, D_OUT);
    pack_f16_kernel<<<packGrid((size_t)RrP * KpIn), 256, 0, stream>>>(
        r_feats, rfeath, Rr, D_IN, RrP, KpIn, D_IN, 1);

    // ---- GEMMs via WMMA (8 waves / block) ----
    const int tilesN    = NoutP / 16;                         // 7
    const int wavesNode = (Nn / 16) * tilesN;                 // 21875
    const int wavesRel  = (RrP / 16) * tilesN;                // 91
    auto gemmGrid = [](int waves) { return (waves + 7) / 8; };

    wmma_gemm_f16<<<gemmGrid(wavesRel), 256, 0, stream>>>(
        wcomph, relatth, nullptr, relw, Rr, D_OUT, KpB, wavesRel);
    wmma_gemm_f16<<<gemmGrid(wavesNode), 256, 0, stream>>>(
        Xh, Wkh, Wk_b, kbuf, Nn, D_OUT, KpIn, wavesNode);
    wmma_gemm_f16<<<gemmGrid(wavesNode), 256, 0, stream>>>(
        Xh, Wqh, Wq_b, qbuf, Nn, D_OUT, KpIn, wavesNode);
    wmma_gemm_f16<<<gemmGrid(wavesNode), 256, 0, stream>>>(
        Xh, Wvh, Wv_b, vbuf, Nn, D_OUT, KpIn, wavesNode);
    wmma_gemm_f16<<<gemmGrid(wavesNode), 256, 0, stream>>>(
        Xh, Wsh, Ws_b, sbuf, Nn, D_OUT, KpIn, wavesNode);
    wmma_gemm_f16<<<gemmGrid(wavesRel), 256, 0, stream>>>(
        rfeath, Wrh, Wr_b, out + ND, Rr, D_OUT, KpIn, wavesRel);

    // ---- edge phase ----
    const int waveBlocks = (Ee * 32 + 255) / 256;
    const int thrBlocks  = (Ee + 255) / 256;
    edge_att_kernel<<<waveBlocks, 256, 0, stream>>>(kbuf, qbuf, relw, src, dst, etype, att, Ee);
    edge_max_kernel<<<thrBlocks, 256, 0, stream>>>(att, dst, segmax, Ee);
    edge_exp_kernel<<<thrBlocks, 256, 0, stream>>>(att, dst, segmax, ex, denom, Ee);
    edge_scatter_kernel<<<waveBlocks, 256, 0, stream>>>(ex, denom, vbuf, src, dst, fbuf, Ee);

    // ---- gated combine + batchnorm + tanh ----
    const size_t total = ND;
    const int elBlocks = (int)((total + 255) / 256);
    combine_kernel<<<elBlocks, 256, 0, stream>>>(sbuf, fbuf, alpha, total);
    bn_stats_kernel<<<D_OUT, 256, 0, stream>>>(fbuf, stats);
    bn_tanh_kernel<<<elBlocks, 256, 0, stream>>>(fbuf, stats, bn_gamma, bn_beta, out, total);

    (void)in_sizes; (void)n_in; (void)out_size; (void)ws_size;
}
